// SoftPixelCNN_62629213110356
// MI455X (gfx1250) — matrix operand order repaired and checked by reading, not compiled, not run on hardware
//
#include <hip/hip_runtime.h>
#include <hip/hip_bf16.h>

// SoftPixelCNN KNN-accumulate for MI455X (gfx1250, wave32).
//
// out[v, b*32 + f] = (1/32) * sum_k exp(-10*||c_v - c_{n(v,k)}||^2) * feat[n(v,k), f]
// for b = 0..6 (the 7 offsets cancel in the pairwise distance, so all blocks equal).
//
// One wave per vertex. The 1x32 (weights) x 32x32 (gathered features) matvec is
// done with two v_wmma_f32_16x16x32_f16 ops; the weight vector is replicated
// into all 16 A-rows so every C row equals the answer (no shuffle needed on
// output). Output is written with non-temporal stores so the 134 MB streamed
// result does not evict the L2-resident 19 MB feature table.

typedef __attribute__((ext_vector_type(16))) _Float16 v16h;
typedef __attribute__((ext_vector_type(8)))  float    v8f;

#define WAVES_PER_BLOCK 8
#define KNN 32
#define NFEAT 32
#define NOFF 7   // 2*D + 1

__global__ __launch_bounds__(256)
void softpixel_knn_wmma_kernel(const float* __restrict__ coords,   // (V,3)
                               const float* __restrict__ feats,    // (V,32)
                               const int*   __restrict__ nidx,     // (V,32)
                               float*       __restrict__ out,      // (V,224)
                               int V)
{
    __shared__ int   s_idx[WAVES_PER_BLOCK][KNN];
    __shared__ float s_w  [WAVES_PER_BLOCK][KNN];

    const int lane = threadIdx.x & 31;
    const int wave = threadIdx.x >> 5;
    const int v    = blockIdx.x * WAVES_PER_BLOCK + wave;   // wave-uniform
    if (v >= V) return;                                     // whole wave exits together

    // ---- per-lane neighbor: lane k handles neighbour slot k -----------------
    const int nk = nidx[(size_t)v * KNN + lane];
    s_idx[wave][lane] = nk;

    // center coords (wave-uniform, L2-hot) and neighbor coords (gather)
    const float cx = coords[(size_t)v * 3 + 0];
    const float cy = coords[(size_t)v * 3 + 1];
    const float cz = coords[(size_t)v * 3 + 2];
    const float dx = cx - coords[(size_t)nk * 3 + 0];
    const float dy = cy - coords[(size_t)nk * 3 + 1];
    const float dz = cz - coords[(size_t)nk * 3 + 2];
    const float d2 = dx * dx + dy * dy + dz * dz;
    // exp(-10*d2) = exp2(-10*log2(e)*d2) -> single v_exp_f32 (TRANS)
    const float w = __builtin_exp2f(-14.426950408889634f * d2);
    s_w[wave][lane] = w;

    // Intra-wave LDS is in-order on CDNA5, but make the cross-lane handoff
    // explicit with the split dependency counter (also fences the idx stores).
    asm volatile("s_wait_dscnt 0x0" ::: "memory");

    // ---- build A (16x32 f16): every row = weight vector w -------------------
    // Layout (16-bit A 16x32, matching B): lane m holds K=0..15 (elts 0..15);
    // lane m+16 holds K=16..31.
    const int abase = (lane & 16) ? 16 : 0;
    v16h A;
#pragma unroll
    for (int e = 0; e < 8; ++e) {
        A[e]     = (_Float16)s_w[wave][abase + e];
        A[e + 8] = (_Float16)s_w[wave][8 + abase + e];
    }

    // ---- build B (32x16 f16) for feature cols 0..15 and 16..31 --------------
    // Layout: lanes 0..15 = column n (K=0..15), lanes 16..31 = column n (K=16..31);
    // VGPR r packs K = kbase+2r (lo half) and kbase+2r+1 (hi half).
    const int col   = lane & 15;
    const int kbase = (lane & 16) ? 16 : 0;
    v16h Blo, Bhi;
#pragma unroll
    for (int r = 0; r < 8; ++r) {
        const int k0 = kbase + 2 * r;
        const int i0 = s_idx[wave][k0];
        const int i1 = s_idx[wave][k0 + 1];
        // coalesced: 16 lanes read 64 contiguous bytes of one feature row
        Blo[2 * r]     = (_Float16)feats[(size_t)i0 * NFEAT + col];
        Blo[2 * r + 1] = (_Float16)feats[(size_t)i1 * NFEAT + col];
        Bhi[2 * r]     = (_Float16)feats[(size_t)i0 * NFEAT + 16 + col];
        Bhi[2 * r + 1] = (_Float16)feats[(size_t)i1 * NFEAT + 16 + col];
    }

    // ---- D = A x B : two 16x16x32 f16 WMMAs (f32 accumulate) -----------------
    v8f clo = {};
    v8f chi = {};
    clo = __builtin_amdgcn_wmma_f32_16x16x32_f16(false, A, false, Blo,
                                                 (short)0, clo, false, false);
    chi = __builtin_amdgcn_wmma_f32_16x16x32_f16(false, A, false, Bhi,
                                                 (short)0, chi, false, false);

    // All C rows are identical (all A rows equal), so:
    //   clo VGPR0, lanes 0..15  -> feature n = lane        (row M=0)
    //   chi VGPR0, lanes 16..31 -> feature 16 + (lane-16)  (row M=8 == row 0)
    const float val = ((lane < 16) ? clo[0] : chi[0]) * (1.0f / (float)KNN);

    // ---- write the 7 identical output blocks (NT: don't pollute L2) ----------
    float* orow = out + (size_t)v * (NOFF * NFEAT);
#pragma unroll
    for (int b = 0; b < NOFF; ++b) {
        __builtin_nontemporal_store(val, orow + b * NFEAT + lane);
    }
}

extern "C" void kernel_launch(void* const* d_in, const int* in_sizes, int n_in,
                              void* d_out, int out_size, void* d_ws, size_t ws_size,
                              hipStream_t stream) {
    const float* coords = (const float*)d_in[0];   // (V,3) f32
    const float* feats  = (const float*)d_in[1];   // (V,32) f32
    const int*   nidx   = (const int*)d_in[2];     // (V,32) i32
    float*       out    = (float*)d_out;           // (V,224) f32

    const int V = in_sizes[0] / 3;
    const int blocks = (V + WAVES_PER_BLOCK - 1) / WAVES_PER_BLOCK;
    softpixel_knn_wmma_kernel<<<blocks, WAVES_PER_BLOCK * 32, 0, stream>>>(
        coords, feats, nidx, out, V);
}
